// GAT_79379585565145
// MI455X (gfx1250) — matrix operand order repaired and checked by reading, not compiled, not run on hardware
//
#include <hip/hip_runtime.h>
#include <hip/hip_fp16.h>

typedef __attribute__((ext_vector_type(16))) _Float16 v16h;
typedef __attribute__((ext_vector_type(8)))  float    v8f;
typedef __attribute__((ext_vector_type(4)))  unsigned int u32x4;
typedef __attribute__((ext_vector_type(8)))  int      i32x8;
typedef __attribute__((ext_vector_type(4)))  int      i32x4;

#define LRELU_ALPHA 0.2f
#define NEG_BIG    -9.0e15f
#define FLT_LOW    -3.0e38f

union FragAB {           // 16 halves = 8 VGPRs (A or B operand of 16x16x32 f16 WMMA)
    v16h  v;
    uint4 u[2];
    _Float16 e[16];
};
union Half8 {            // 8 halves = one b128 store
    uint4 u;
    _Float16 e[8];
};

// LDS tile: 64 rows x 32 halves, padded by TDM to 40 halves (80 B) per row.
#define TILE_PITCH_H 40

// ---------------------------------------------------------------------------
// TDM: load one 64-row x 32-col f16 tile (row pitch 1024 halves in memory)
// into LDS at lds_addr with 16 B of padding inserted after every 64 B row.
// D# per CDNA5 ISA ch.8; wave-uniform; tracked with TENSORcnt.
// ---------------------------------------------------------------------------
__device__ __forceinline__ void tdm_load_tile(const _Float16* gsrc,
                                              unsigned lds_addr) {
    unsigned long long ga = (unsigned long long)(uintptr_t)gsrc;
    u32x4 g0;
    g0[0] = 1u;                                        // count=1, user desc
    g0[1] = lds_addr;                                  // lds_addr (bytes)
    g0[2] = (unsigned)(ga & 0xffffffffu);              // global_addr[31:0]
    g0[3] = (unsigned)((ga >> 32) & 0x01ffffffu)       // global_addr[56:32]
            | 0x80000000u;                             // type=2 ("image")
    i32x8 g1;
    g1[0] = (int)((1u << 16)        // data_size = 2 bytes
                | (1u << 20)        // pad_enable
                | (3u << 22)        // pad_interval: 16 dwords (64 B)
                | (3u << 25));      // pad_amount:   4 dwords (16 B)
    g1[1] = (int)(1024u << 16);     // tensor_dim0 = 1024 (low 16)
    g1[2] = (int)(64u << 16);       // tensor_dim0 hi = 0; tensor_dim1 = 64
    g1[3] = (int)(32u << 16);       // tensor_dim1 hi = 0; tile_dim0 = 32
    g1[4] = 64;                     // tile_dim1 = 64; tile_dim2 = 0
    g1[5] = 1024;                   // tensor_dim0_stride (low 32)
    g1[6] = 0;
    g1[7] = 0;
    i32x4 gz = {0, 0, 0, 0};
#if defined(__clang_major__) && (__clang_major__ >= 23)
    i32x8 gz8 = {0, 0, 0, 0, 0, 0, 0, 0};
    __builtin_amdgcn_tensor_load_to_lds(g0, g1, gz, gz, gz8, 0);
#else
    __builtin_amdgcn_tensor_load_to_lds(g0, g1, gz, gz, 0);
#endif
}

// ---------------------------------------------------------------------------
// 1) Pack x[:, :16] (fp32, [2,17,1024,64]) -> Xh (f16, [32,1024,64])
// ---------------------------------------------------------------------------
__global__ __launch_bounds__(256) void pack_x(const float* __restrict__ x,
                                              _Float16* __restrict__ Xh) {
    int idx = blockIdx.x * 256 + threadIdx.x;          // 0 .. 32*1024*64-1
    int bt  = idx >> 16;                               // 1024*64 = 65536
    int rem = idx & 65535;
    int b = bt >> 4, t = bt & 15;
    Xh[idx] = (_Float16)x[((size_t)(b * 17 + t) << 16) + rem];
}

// ---------------------------------------------------------------------------
// 2) Convert + transpose W [K,64] fp32 -> Wt [64][K] f16
// ---------------------------------------------------------------------------
__global__ __launch_bounds__(256) void conv_wt(const float* __restrict__ W,
                                               _Float16* __restrict__ Wt, int K) {
    int idx = blockIdx.x * 256 + threadIdx.x;
    if (idx >= 64 * K) return;
    int n = idx / K, k = idx - n * K;
    Wt[idx] = (_Float16)W[(size_t)k * 64 + n];
}

// ---------------------------------------------------------------------------
// 3) WMMA GEMM: H[m, 0..63] = A[m, :] @ W  (W transposed, [64][K]).
//    Side products per row: f1[m] = H[m,:]·a1, f2[m] = H[m,:]·a2.
//    H stored TRANSPOSED f16 at Ht[(bt*64 + n)*1024 + (m & 1023)].
//    Wave: 16 rows x 64 cols; block = 4 waves; M = 32768.
// ---------------------------------------------------------------------------
__global__ __launch_bounds__(128) void gemm_rowstats(
    const _Float16* __restrict__ A, int K,
    const _Float16* __restrict__ Wt,
    const float* __restrict__ a1, const float* __restrict__ a2,
    _Float16* __restrict__ Ht,
    float* __restrict__ F1, float* __restrict__ F2) {

    const int lane = threadIdx.x & 31;
    const int wave = threadIdx.x >> 5;
    const int lm = lane & 15;
    const int hi = lane >> 4;
    const int m0 = (blockIdx.x * 4 + wave) * 16;
    const int mA = m0 + lm;

    v8f acc[4] = {};
    const _Float16* arow = A + (size_t)mA * K;

    for (int k0 = 0; k0 < K; k0 += 32) {
        FragAB a;
        // A 16x32 f16 layout: half i -> k = 16*(i>>3) + 8*hi + (i&7)
        a.u[0] = *reinterpret_cast<const uint4*>(arow + k0 + 8 * hi);
        a.u[1] = *reinterpret_cast<const uint4*>(arow + k0 + 16 + 8 * hi);
#pragma unroll
        for (int nt = 0; nt < 4; ++nt) {
            FragAB b;
            // B 32x16 f16 layout: half i -> k = 16*(i>>3) + 8*hi + (i&7), col = lm
            const _Float16* bp = Wt + (size_t)(nt * 16 + lm) * K + k0;
            b.u[0] = *reinterpret_cast<const uint4*>(bp + 8 * hi);
            b.u[1] = *reinterpret_cast<const uint4*>(bp + 16 + 8 * hi);
            acc[nt] = __builtin_amdgcn_wmma_f32_16x16x32_f16(
                false, a.v, false, b.v, (short)0, acc[nt], false, false);
        }
    }

    float a1v[4], a2v[4];
#pragma unroll
    for (int nt = 0; nt < 4; ++nt) {
        a1v[nt] = a1[nt * 16 + lm];
        a2v[nt] = a2[nt * 16 + lm];
    }
#pragma unroll
    for (int r = 0; r < 8; ++r) {
        float s1 = 0.f, s2 = 0.f;
#pragma unroll
        for (int nt = 0; nt < 4; ++nt) {
            s1 += acc[nt][r] * a1v[nt];
            s2 += acc[nt][r] * a2v[nt];
        }
#pragma unroll
        for (int off = 1; off < 16; off <<= 1) {
            s1 += __shfl_xor(s1, off, 32);
            s2 += __shfl_xor(s2, off, 32);
        }
        if (lm == 0) {
            F1[m0 + 8 * hi + r] = s1;
            F2[m0 + 8 * hi + r] = s2;
        }
    }

    const int bt = m0 >> 10;
    const int ml = (m0 & 1023) + 8 * hi;
#pragma unroll
    for (int nt = 0; nt < 4; ++nt) {
        Half8 hh;
#pragma unroll
        for (int r = 0; r < 8; ++r) hh.e[r] = (_Float16)acc[nt][r];
        *reinterpret_cast<uint4*>(
            Ht + ((size_t)bt * 64 + nt * 16 + lm) * 1024 + ml) = hh.u;
    }
}

// ---------------------------------------------------------------------------
// 4) Fused masked-softmax attention (flash style) + WMMA aggregation.
//    H tiles staged into LDS by the Tensor Data Mover (double-buffered),
//    shared by all 4 waves; f2 row staged once in LDS.
//    mode 0: ELU -> hcat f16; mode 1: ReLU -> out f32
// ---------------------------------------------------------------------------
__global__ __launch_bounds__(128) void gat_attn(
    const float* __restrict__ F1, const float* __restrict__ F2,
    const _Float16* __restrict__ Ht, const int* __restrict__ adj,
    float* __restrict__ outF, _Float16* __restrict__ outH, int mode) {

    __shared__ __align__(16) _Float16 ltile[2][64 * TILE_PITCH_H]; // 2 x 5 KB
    __shared__ __align__(16) float    f2s[1024];                   // 4 KB

    const int lane = threadIdx.x & 31;
    const int wave = threadIdx.x >> 5;
    const int lm = lane & 15;
    const int hi = lane >> 4;
    const int bt = blockIdx.y;
    const int h  = blockIdx.z;
    const int i0 = (blockIdx.x * 4 + wave) * 16;

    const int grp = h * 32 + bt;
    const float*    f1p = F1 + (size_t)grp * 1024;
    const float*    f2p = F2 + (size_t)grp * 1024;
    const _Float16* Hb  = Ht + (size_t)grp * 65536;   // [64][1024] f16

    // Kick off the TDM for tile 0 immediately, then stage f2 cooperatively.
    if (wave == 0)
        tdm_load_tile(Hb, (unsigned)(uintptr_t)&ltile[0][0]);
    {
        const float4* src = reinterpret_cast<const float4*>(f2p);
        float4*       dst = reinterpret_cast<float4*>(f2s);
#pragma unroll
        for (int t = 0; t < 2; ++t)
            dst[threadIdx.x + 128 * t] = src[threadIdx.x + 128 * t];
    }

    const int   irow = i0 + lm;
    const float f1i  = f1p[irow];
    const int*  arow = adj + (size_t)irow * 1024;

    float mrun = FLT_LOW, lrun = 0.f;
    v8f acc[4] = {};
    int buf = 0;

    for (int j0 = 0; j0 < 1024; j0 += 32, buf ^= 1) {
        // Tile j0 resident after this wait+barrier; barrier also proves every
        // wave finished reading buf^1 last iteration, so it is safe to refill.
        if (wave == 0)
            __builtin_amdgcn_s_wait_tensorcnt(0);
        __syncthreads();
        if (wave == 0 && j0 + 32 < 1024)
            tdm_load_tile(Hb + j0 + 32,
                          (unsigned)(uintptr_t)&ltile[buf ^ 1][0]);

        // ---- scores in exact A-frag order: j = j0 + 16g + 8hi + t ----
        float s16[16];
        float tmax = FLT_LOW;
#pragma unroll
        for (int g = 0; g < 2; ++g) {
            const int jb = j0 + 16 * g + 8 * hi;
            float4 fa = *reinterpret_cast<const float4*>(&f2s[jb]);
            float4 fb = *reinterpret_cast<const float4*>(&f2s[jb + 4]);
            int4   ia = *reinterpret_cast<const int4*>(arow + jb);
            int4   ib = *reinterpret_cast<const int4*>(arow + jb + 4);
            float fv[8] = {fa.x, fa.y, fa.z, fa.w, fb.x, fb.y, fb.z, fb.w};
            int   av[8] = {ia.x, ia.y, ia.z, ia.w, ib.x, ib.y, ib.z, ib.w};
#pragma unroll
            for (int t = 0; t < 8; ++t) {
                float s = f1i + fv[t];
                s = (s > 0.f) ? s : LRELU_ALPHA * s;       // leaky_relu
                s = (av[t] != 0) ? s : NEG_BIG;            // adjacency mask
                s16[g * 8 + t] = s;
                tmax = fmaxf(tmax, s);
            }
        }
        if (j0 + 32 < 1024)
            __builtin_prefetch(arow + j0 + 32, 0, 1);

        // ---- online softmax statistics (row split across lane pair) ----
        tmax = fmaxf(tmax, __shfl_xor(tmax, 16, 32));
        const float mnew  = fmaxf(mrun, tmax);
        const float scale = __expf(mrun - mnew);
        float rsum = 0.f;
        FragAB pf;
#pragma unroll
        for (int t = 0; t < 16; ++t) {
            float pv = __expf(s16[t] - mnew);
            rsum += pv;
            pf.e[t] = (_Float16)pv;
        }
        rsum += __shfl_xor(rsum, 16, 32);
        lrun = lrun * scale + rsum;
        mrun = mnew;

        // ---- rescale accumulator (per-row scale via ds_bpermute) ----
#pragma unroll
        for (int r = 0; r < 8; ++r) {
            float sc = __shfl(scale, 8 * hi + r, 32);
#pragma unroll
            for (int nt = 0; nt < 4; ++nt) acc[nt][r] = acc[nt][r] * sc;
        }

        // ---- P (16x32 f16) @ H-tile (32x64 f16): B-frags from LDS ----
#pragma unroll
        for (int nt = 0; nt < 4; ++nt) {
            FragAB b;
            const _Float16* bp =
                &ltile[buf][(nt * 16 + lm) * TILE_PITCH_H];
            b.u[0] = *reinterpret_cast<const uint4*>(bp + 8 * hi);
            b.u[1] = *reinterpret_cast<const uint4*>(bp + 16 + 8 * hi);
            acc[nt] = __builtin_amdgcn_wmma_f32_16x16x32_f16(
                false, pf.v, false, b.v, (short)0, acc[nt], false, false);
        }
    }

    // ---- epilogue: normalize, activate, store ----
#pragma unroll
    for (int r = 0; r < 8; ++r) {
        float lr   = __shfl(lrun, 8 * hi + r, 32);
        float linv = 1.0f / lr;
        int   row  = i0 + 8 * hi + r;
#pragma unroll
        for (int nt = 0; nt < 4; ++nt) {
            float v = acc[nt][r] * linv;
            if (mode == 0) {
                v = (v > 0.f) ? v : (__expf(v) - 1.0f);    // ELU
                outH[((size_t)bt * 1024 + row) * 256 + h * 64 + nt * 16 + lm] =
                    (_Float16)v;
            } else {
                v = (v > 0.f) ? v : 0.f;                   // ReLU
                outF[((size_t)bt * 1024 + row) * 64 + nt * 16 + lm] = v;
            }
        }
    }
}

// ---------------------------------------------------------------------------
extern "C" void kernel_launch(void* const* d_in, const int* in_sizes, int n_in,
                              void* d_out, int out_size, void* d_ws, size_t ws_size,
                              hipStream_t stream) {
    const float* x       = (const float*)d_in[0];   // [2,17,1024,64]
    const int*   adj     = (const int*)d_in[1];     // [1024,1024]
    const float* W_heads = (const float*)d_in[2];   // [4,64,64]
    const float* a_heads = (const float*)d_in[3];   // [4,128,1]
    const float* W_out   = (const float*)d_in[4];   // [256,64]
    const float* a_out   = (const float*)d_in[5];   // [128,1]
    float*       out     = (float*)d_out;           // [2,16,1024,64] fp32

    char* w = (char*)d_ws;
    auto alloc = [&](size_t bytes) -> char* {
        char* p = w;
        w += (bytes + 255) & ~(size_t)255;
        return p;
    };
    _Float16* Xh   = (_Float16*)alloc(32ull * 1024 * 64 * 2);       // 4 MiB
    _Float16* WtH  = (_Float16*)alloc(4ull * 64 * 64 * 2);
    _Float16* WtO  = (_Float16*)alloc(64ull * 256 * 2);
    _Float16* Ht1  = (_Float16*)alloc(4ull * 32 * 64 * 1024 * 2);   // 16 MiB
    float*    F1a  = (float*)alloc(4ull * 32 * 1024 * 4);
    float*    F2a  = (float*)alloc(4ull * 32 * 1024 * 4);
    _Float16* hcat = (_Float16*)alloc(32ull * 1024 * 256 * 2);      // 16 MiB
    _Float16* Ht2  = (_Float16*)alloc(32ull * 64 * 1024 * 2);       // 4 MiB
    float*    F1b  = (float*)alloc(32ull * 1024 * 4);
    float*    F2b  = (float*)alloc(32ull * 1024 * 4);

    // 1) pack + convert inputs
    pack_x<<<8192, 256, 0, stream>>>(x, Xh);
    for (int h = 0; h < 4; ++h)
        conv_wt<<<(64 * 64 + 255) / 256, 256, 0, stream>>>(
            W_heads + (size_t)h * 64 * 64, WtH + (size_t)h * 64 * 64, 64);
    conv_wt<<<(64 * 256 + 255) / 256, 256, 0, stream>>>(W_out, WtO, 256);

    // 2) layer-1: per-head H = X@W, row scores, transposed H
    for (int h = 0; h < 4; ++h)
        gemm_rowstats<<<512, 128, 0, stream>>>(
            Xh, 64, WtH + (size_t)h * 64 * 64,
            a_heads + (size_t)h * 128, a_heads + (size_t)h * 128 + 64,
            Ht1 + (size_t)h * 32 * 64 * 1024,
            F1a + (size_t)h * 32 * 1024, F2a + (size_t)h * 32 * 1024);

    // 3) layer-1 fused attention -> ELU -> hcat (f16, [32*1024, 256])
    gat_attn<<<dim3(16, 32, 4), 128, 0, stream>>>(
        F1a, F2a, Ht1, adj, nullptr, hcat, 0);

    // 4) layer-2: H2 = hcat @ W_out (K=256), scores, transposed H2
    gemm_rowstats<<<512, 128, 0, stream>>>(
        hcat, 256, WtO, a_out, a_out + 64, Ht2, F1b, F2b);

    // 5) layer-2 fused attention -> ReLU -> fp32 output
    gat_attn<<<dim3(16, 32, 1), 128, 0, stream>>>(
        F1b, F2b, Ht2, adj, out, nullptr, 1);
}
